// TorchAttractorLanguageModel_63660005261847
// MI455X (gfx1250) — matrix operand order repaired and checked by reading, not compiled, not run on hardware
//
#include <hip/hip_runtime.h>
#include <hip/hip_bf16.h>
#include <math.h>

#define VOCAB 50257
#define DIM   512
#define NHEAD 4
#define RANK  64
#define SEQL  512
#define DTSTEP 0.05f
#define NCONV 12
#define NATTR 16

#define HR    (NHEAD*RANK)   // 256
#define KB1   (DIM/32)       // 16 K-blocks, GEMM1 (K=512)
#define NT1   (HR/16)        // 16 N-tiles,  GEMM1 (N=256)
#define KB2   (HR/32)        // 8  K-blocks, GEMM2 (K=256)
#define NT2   (DIM/16)       // 32 N-tiles,  GEMM2 (N=512)

typedef __attribute__((ext_vector_type(16))) _Float16 v16h;
typedef __attribute__((ext_vector_type(8)))  float    v8f;

__device__ __forceinline__ v8f zero8() {
  v8f z; 
#pragma unroll
  for (int i = 0; i < 8; ++i) z[i] = 0.0f;
  return z;
}

__device__ __forceinline__ v8f wmma16(v16h a, v16h b, v8f c) {
  // D = A(16x32 f16) * B(32x16 f16) + C(16x16 f32)
  return __builtin_amdgcn_wmma_f32_16x16x32_f16(false, a, false, b, (short)0, c,
                                                false, false);
}

// A-matrix (16x32, f16) lane layout per CDNA5 ISA 7.12.2:
// lanes 0-15 row M=lane; VGPR j<4 -> K=2j,2j+1 ; j>=4 -> K=16+2(j-4),+1
// lanes 16-31: same rows, K offset +8.
__device__ __forceinline__ v16h load_a_frag(const _Float16* __restrict__ buf,
                                            int rowlen, int m, int kb, int hi8) {
  const _Float16* r = buf + m * rowlen + kb * 32 + hi8;
  v16h a;
#pragma unroll
  for (int j = 0; j < 4; ++j) { a[2*j] = r[2*j]; a[2*j+1] = r[2*j+1]; }
#pragma unroll
  for (int j = 4; j < 8; ++j) { int k = 16 + 2*(j-4); a[2*j] = r[k]; a[2*j+1] = r[k+1]; }
  return a;
}

// ---- weight packing into fragment-native B layout -------------------------
// B(32x16) lane layout: lane -> N = lane%16 ; elem i -> K = (i&7) + (i>=8 ? 16 : 0) + (lane>=16 ? 8 : 0).
// Packed so a fragment is 16 contiguous halfs per lane (one 32B load).

// GEMM1 B: B[K=d][N=hr] = Vm[h][r][d]   (Vm is (H,R,D) row-major)
__global__ void pack_wv(const float* __restrict__ Vm, _Float16* __restrict__ WvP) {
  int t = blockIdx.x * 256 + threadIdx.x;            // 16*16*512 = 131072
  int i = t & 15, l = (t >> 4) & 31, nt = (t >> 9) & 15, kb = t >> 13;
  int d  = kb * 32 + ((l >= 16) ? 8 : 0) + (i & 7) + ((i >= 8) ? 16 : 0);
  int hr = nt * 16 + (l & 15);
  WvP[t] = (_Float16)Vm[(size_t)hr * DIM + d];
}

// GEMM2 B: B[K=hr][N=d] = U[h][d][r]    (U is (H,D,R) row-major)
__global__ void pack_wu(const float* __restrict__ U, _Float16* __restrict__ WuP) {
  int t = blockIdx.x * 256 + threadIdx.x;            // 8*32*512 = 131072
  int i = t & 15, l = (t >> 4) & 31, nt = (t >> 9) & 31, kb = t >> 14;
  int hr = kb * 32 + ((l >= 16) ? 8 : 0) + (i & 7) + ((i >= 8) ? 16 : 0);
  int d  = nt * 16 + (l & 15);
  int h = hr >> 6, r = hr & 63;
  WuP[t] = (_Float16)U[((size_t)h * DIM + d) * RANK + r];
}

// f32 transpose of U for the sequential scan: UT[d][h*64+r] = U[h][d][r]
__global__ void pack_ut(const float* __restrict__ U, float* __restrict__ UT) {
  int t = blockIdx.x * 256 + threadIdx.x;            // 512*256 = 131072
  int r = t & 63, h = (t >> 6) & 3, d = t >> 8;
  UT[t] = U[((size_t)h * DIM + d) * RANK + r];
}

// ---- Phase A: converge all vocab signals from zero (NA Euler steps) -------
__global__ void __launch_bounds__(256)
attractor_kernel(const float* __restrict__ embed,
                 const _Float16* __restrict__ WvP,
                 const _Float16* __restrict__ WuP,
                 float* __restrict__ attr,
                 float* __restrict__ asq,
                 const float* __restrict__ cubic_p,
                 const float* __restrict__ coup_p) {
  __shared__ float    s32 [16 * DIM];
  __shared__ float    sg32[16 * DIM];
  __shared__ _Float16 s16 [16 * DIM];
  __shared__ _Float16 t16 [16 * HR];
  __shared__ float    psum[256];

  const int tid  = threadIdx.x;
  const int wave = tid >> 5;
  const int lane = tid & 31;
  const int lo   = lane & 15;
  const int hi8  = (lane >> 4) * 8;
  const int base = blockIdx.x * 16;

  const float cub  = cubic_p[0];
  const float coup = coup_p[0];

  for (int idx = tid; idx < 16 * DIM; idx += 256) {
    int mm = idx >> 9, dd = idx & (DIM - 1);
    int row = base + mm;
    s32[idx] = 0.0f;
    s16[idx] = (_Float16)0.0f;
    sg32[idx] = (row < VOCAB) ? embed[(size_t)row * DIM + dd] : 0.0f;
  }
  __syncthreads();

  const int nt1a = wave * 2;   // 2 GEMM1 output tiles per wave (16 total)
  const int nt2a = wave * 4;   // 4 GEMM2 output tiles per wave (32 total)

  for (int it = 0; it < NATTR; ++it) {
    // GEMM1: pre(16xHR) = s16(16xDIM) * WvP(DIMxHR)
    v8f p0 = zero8(), p1 = zero8();
#pragma unroll 4
    for (int kb = 0; kb < KB1; ++kb) {
      v16h a  = load_a_frag(s16, DIM, lo, kb, hi8);
      v16h b0 = *(const v16h*)(WvP + (size_t)(kb * NT1 + nt1a    ) * 512 + lane * 16);
      v16h b1 = *(const v16h*)(WvP + (size_t)(kb * NT1 + nt1a + 1) * 512 + lane * 16);
      p0 = wmma16(a, b0, p0);
      p1 = wmma16(a, b1, p1);
    }
    // tanh -> t16[16 x HR] (D-tile: row = j + 8*(lane>=16), col = lane%16)
#pragma unroll
    for (int j = 0; j < 8; ++j) {
      int mr = j + hi8;
      t16[mr * HR + (nt1a    ) * 16 + lo] = (_Float16)tanhf(p0[j]);
      t16[mr * HR + (nt1a + 1) * 16 + lo] = (_Float16)tanhf(p1[j]);
    }
    __syncthreads();

    // GEMM2: low(16xDIM) = t16(16xHR) * WuP(HRxDIM)
    v8f l0 = zero8(), l1 = zero8(), l2 = zero8(), l3 = zero8();
#pragma unroll 2
    for (int kb = 0; kb < KB2; ++kb) {
      v16h a  = load_a_frag(t16, HR, lo, kb, hi8);
      v16h b0 = *(const v16h*)(WuP + (size_t)(kb * NT2 + nt2a    ) * 512 + lane * 16);
      v16h b1 = *(const v16h*)(WuP + (size_t)(kb * NT2 + nt2a + 1) * 512 + lane * 16);
      v16h b2 = *(const v16h*)(WuP + (size_t)(kb * NT2 + nt2a + 2) * 512 + lane * 16);
      v16h b3 = *(const v16h*)(WuP + (size_t)(kb * NT2 + nt2a + 3) * 512 + lane * 16);
      l0 = wmma16(a, b0, l0);
      l1 = wmma16(a, b1, l1);
      l2 = wmma16(a, b2, l2);
      l3 = wmma16(a, b3, l3);
    }
    // Euler update (each (m,d) owned by exactly one lane)
#pragma unroll
    for (int j = 0; j < 8; ++j) {
      int mr = j + hi8;
#pragma unroll
      for (int q = 0; q < 4; ++q) {
        float lw = (q == 0) ? l0[j] : (q == 1) ? l1[j] : (q == 2) ? l2[j] : l3[j];
        int idx = mr * DIM + (nt2a + q) * 16 + lo;
        float sv = s32[idx];
        float ns = sv + DTSTEP * (sg32[idx] - sv - cub * sv * sv * sv + coup * lw);
        s32[idx] = ns;
        s16[idx] = (_Float16)ns;
      }
    }
    __syncthreads();
  }

  // write attractors + per-row sum of squares
  for (int idx = tid; idx < 16 * DIM; idx += 256) {
    int mm = idx >> 9, dd = idx & (DIM - 1);
    int row = base + mm;
    if (row < VOCAB) attr[(size_t)row * DIM + dd] = s32[idx];
  }
  {
    int mm = tid >> 4, c = tid & 15;
    float acc = 0.0f;
    for (int dd = c * 32; dd < c * 32 + 32; ++dd) {
      float v = s32[mm * DIM + dd];
      acc += v * v;
    }
    psum[tid] = acc;
  }
  __syncthreads();
  if (tid < 16) {
    float acc = 0.0f;
    for (int c = 0; c < 16; ++c) acc += psum[tid * 16 + c];
    int row = base + tid;
    if (row < VOCAB) asq[row] = acc;
  }
}

// ---- Phase B: sequential teacher-forced scan (latency-bound recurrence) ---
__global__ void __launch_bounds__(512)
scan_kernel(const float* __restrict__ embed,
            const float* __restrict__ VmRM,   // (HR x DIM) row-major == Vm input
            const float* __restrict__ UT,     // (DIM x HR) row-major
            const int* __restrict__ ids,
            const float* __restrict__ cubic_p,
            const float* __restrict__ coup_p,
            float* __restrict__ states) {
  __shared__ float s[DIM];
  __shared__ float t[HR];
  const int tid = threadIdx.x;
  const float cub = cubic_p[0], coup = coup_p[0];
  s[tid] = 0.0f;
  __syncthreads();
  for (int l = 0; l < SEQL; ++l) {
    const float sigv = embed[(size_t)ids[l] * DIM + tid];
    for (int it = 0; it < NCONV; ++it) {
      if (tid < HR) {
        const float* vr = VmRM + (size_t)tid * DIM;
        float acc = 0.0f;
#pragma unroll 8
        for (int d = 0; d < DIM; ++d) acc += vr[d] * s[d];
        t[tid] = tanhf(acc);
      }
      __syncthreads();
      {
        const float* ur = UT + (size_t)tid * HR;
        float low = 0.0f;
#pragma unroll 8
        for (int k = 0; k < HR; ++k) low += ur[k] * t[k];
        float sv = s[tid];
        s[tid] = sv + DTSTEP * (sigv - sv - cub * sv * sv * sv + coup * low);
      }
      __syncthreads();
    }
    states[(size_t)l * DIM + tid] = s[tid];
  }
}

// ---- Phase C: fused cdist + online log-softmax + NLL (one WG per row) -----
__global__ void __launch_bounds__(256)
nll_kernel(const float* __restrict__ states,
           const float* __restrict__ attr,
           const float* __restrict__ asq,
           const float* __restrict__ temp_p,
           const int* __restrict__ tgt,
           float* __restrict__ nll) {
  __shared__ float sv[DIM];
  __shared__ float rmax[256];
  __shared__ float rsum[256];
  __shared__ float sh_ss;
  __shared__ float sh_tl;
  const int tid = threadIdx.x;
  const int l = blockIdx.x;
  sv[tid]       = states[(size_t)l * DIM + tid];
  sv[tid + 256] = states[(size_t)l * DIM + tid + 256];
  __syncthreads();
  rsum[tid] = sv[tid] * sv[tid] + sv[tid + 256] * sv[tid + 256];
  __syncthreads();
  for (int s2 = 128; s2 > 0; s2 >>= 1) {
    if (tid < s2) rsum[tid] += rsum[tid + s2];
    __syncthreads();
  }
  if (tid == 0) sh_ss = rsum[0];
  __syncthreads();
  const float ss = sh_ss;
  const float inv_tau = 1.0f / fmaxf(temp_p[0], 1e-6f);
  const int target = tgt[l];
  float mx = -3.4e38f;
  float sm = 0.0f;
  for (int v = tid; v < VOCAB; v += 256) {
    const float4* ar = (const float4*)(attr + (size_t)v * DIM);
    const float4* s4 = (const float4*)sv;
    float dot = 0.0f;
#pragma unroll 4
    for (int d = 0; d < DIM / 4; ++d) {
      float4 a4 = ar[d], b4 = s4[d];
      dot += a4.x * b4.x + a4.y * b4.y + a4.z * b4.z + a4.w * b4.w;
    }
    float sq = ss + asq[v] - 2.0f * dot;
    float lg = -sqrtf(fmaxf(sq, 0.0f)) * inv_tau;
    if (v == target) sh_tl = lg;
    if (lg > mx) { sm *= expf(mx - lg); mx = lg; }
    sm += expf(lg - mx);
  }
  rmax[tid] = mx;
  rsum[tid] = sm;
  __syncthreads();
  for (int s2 = 128; s2 > 0; s2 >>= 1) {
    if (tid < s2) {
      float m1 = rmax[tid], m2 = rmax[tid + s2];
      float M = fmaxf(m1, m2);
      rsum[tid] = rsum[tid] * expf(m1 - M) + rsum[tid + s2] * expf(m2 - M);
      rmax[tid] = M;
    }
    __syncthreads();
  }
  if (tid == 0) nll[l] = (rmax[0] + logf(rsum[0])) - sh_tl;
}

__global__ void __launch_bounds__(256)
reduce_kernel(const float* __restrict__ nll, float* __restrict__ out) {
  __shared__ float red[256];
  int tid = threadIdx.x;
  red[tid] = nll[tid] + nll[tid + 256];
  __syncthreads();
  for (int s2 = 128; s2 > 0; s2 >>= 1) {
    if (tid < s2) red[tid] += red[tid + s2];
    __syncthreads();
  }
  if (tid == 0) out[0] = red[0] * (1.0f / SEQL);
}

extern "C" void kernel_launch(void* const* d_in, const int* in_sizes, int n_in,
                              void* d_out, int out_size, void* d_ws, size_t ws_size,
                              hipStream_t stream) {
  const float* embed = (const float*)d_in[0];
  const float* U     = (const float*)d_in[1];
  const float* Vm    = (const float*)d_in[2];
  const float* temp  = (const float*)d_in[3];
  const float* cubic = (const float*)d_in[4];
  const float* coup  = (const float*)d_in[5];
  const int*   ids   = (const int*)d_in[6];
  const int*   tgt   = (const int*)d_in[7];
  float* out = (float*)d_out;
  (void)in_sizes; (void)n_in; (void)out_size; (void)ws_size;

  char* ws = (char*)d_ws;
  size_t off = 0;
  auto take = [&](size_t bytes) {
    char* p = ws + off;
    off = (off + bytes + 255) & ~(size_t)255;
    return p;
  };
  float*    attr   = (float*)    take((size_t)VOCAB * DIM * sizeof(float));
  float*    asq    = (float*)    take((size_t)VOCAB * sizeof(float));
  _Float16* WvP    = (_Float16*) take((size_t)KB1 * NT1 * 512 * sizeof(_Float16));
  _Float16* WuP    = (_Float16*) take((size_t)KB2 * NT2 * 512 * sizeof(_Float16));
  float*    UT     = (float*)    take((size_t)DIM * HR * sizeof(float));
  float*    states = (float*)    take((size_t)SEQL * DIM * sizeof(float));
  float*    nll    = (float*)    take((size_t)SEQL * sizeof(float));

  pack_wv<<<(KB1 * NT1 * 512) / 256, 256, 0, stream>>>(Vm, WvP);
  pack_wu<<<(KB2 * NT2 * 512) / 256, 256, 0, stream>>>(U, WuP);
  pack_ut<<<(DIM * HR) / 256, 256, 0, stream>>>(U, UT);
  attractor_kernel<<<(VOCAB + 15) / 16, 256, 0, stream>>>(embed, WvP, WuP, attr,
                                                          asq, cubic, coup);
  scan_kernel<<<1, 512, 0, stream>>>(embed, Vm, UT, ids, cubic, coup, states);
  nll_kernel<<<SEQL, 256, 0, stream>>>(states, attr, asq, temp, tgt, nll);
  reduce_kernel<<<1, 256, 0, stream>>>(nll, out);
}
